// StageExpertBlock_24446953849148
// MI455X (gfx1250) — matrix operand order repaired and checked by reading, not compiled, not run on hardware
//
#include <hip/hip_runtime.h>
#include <hip/hip_bf16.h>
#include <math.h>

// Problem dims (from reference): B=32 L=256 -> NTOK=8192
#define NTOK 8192
#define DD   256   // D
#define FF   8     // F
#define DFE  64    // DF
#define DIN  320   // D + DF
#define DRR  256   // DR
#define EE   8     // E
#define DHH  512   // DH

typedef __attribute__((ext_vector_type(16))) __bf16 v16bf;
typedef __attribute__((ext_vector_type(8)))  float  v8f;
typedef int v4i __attribute__((vector_size(16)));   // matches builtin's pointee type

union Frag { v16bf v; uint4 q[2]; };

// ---- gfx1250 async global->LDS path (guarded; falls back to vector copy) ----
#if defined(__gfx1250__) && __has_builtin(__builtin_amdgcn_global_load_async_to_lds_b128) && \
    __has_builtin(__builtin_amdgcn_s_wait_asynccnt)
#define HAVE_ASYNC_LDS 1
#else
#define HAVE_ASYNC_LDS 0
#endif

#define GASP __attribute__((address_space(1)))
#define LASP __attribute__((address_space(3)))

__device__ inline void lds_fill(void* lds_dst, const void* gsrc, int n_uint4, int tid, int nthr) {
#if HAVE_ASYNC_LDS
  GASP v4i* g = (GASP v4i*)gsrc;       // C-style cast: drops const + addrspace cast
  LASP v4i* l = (LASP v4i*)lds_dst;
  for (int i = tid; i < n_uint4; i += nthr)
    __builtin_amdgcn_global_load_async_to_lds_b128(g + i, l + i, 0, 0);
  __builtin_amdgcn_s_wait_asynccnt(0);
#else
  const uint4* g = (const uint4*)gsrc;
  uint4* l = (uint4*)lds_dst;
  for (int i = tid; i < n_uint4; i += nthr) l[i] = g[i];
#endif
}

__device__ inline unsigned short f2bf(float f) {
  union { float f; unsigned int u; } v; v.f = f;
  unsigned int u = v.u;
  return (unsigned short)((u + 0x7FFFu + ((u >> 16) & 1u)) >> 16);  // RNE
}
__device__ inline float bf2f(unsigned short h) {
  union { unsigned int u; float f; } v; v.u = ((unsigned int)h) << 16; return v.f;
}
// Branch-free tanh-approx gelu: 0.5x(1+tanh(z)) == x * sigmoid(2z).
// Uses v_exp_f32 / v_rcp_f32 (TRANS ops co-execute with WMMA), no EXEC branches.
__device__ inline float gelu_tanh(float x) {
  float z = 0.7978845608028654f * (x + 0.044715f * x * x * x);
  float s = __builtin_amdgcn_rcpf(1.0f + __expf(-2.0f * z));
  return x * s;
}
// Build a 16-element bf16 fragment from two 16-byte chunks.
__device__ inline v16bf load16(const unsigned short* p0, const unsigned short* p1) {
  Frag f;
  f.q[0] = *(const uint4*)p0;
  f.q[1] = *(const uint4*)p1;
  return f.v;
}
__device__ inline v8f wmma_bf16(v16bf a, v16bf b, v8f c) {
  return __builtin_amdgcn_wmma_f32_16x16x32_bf16(false, a, false, b, (short)0, c, false, false);
}

// ---------------------------------------------------------------------------
// x = concat(hidden, feat @ W_feat + b_feat) cast to bf16.  grid=NTOK, block=DIN
__global__ void k_build_x(const float* __restrict__ hidden, const float* __restrict__ feat,
                          const float* __restrict__ W_feat, const float* __restrict__ b_feat,
                          unsigned short* __restrict__ xbf) {
  int t = blockIdx.x;
  int c = threadIdx.x;  // 0..319
  float v;
  if (c < DD) {
    v = hidden[(size_t)t * DD + c];
  } else {
    int j = c - DD;
    v = b_feat[j];
#pragma unroll
    for (int f = 0; f < FF; ++f) v += feat[(size_t)t * FF + f] * W_feat[f * DFE + j];
  }
  xbf[(size_t)t * DIN + c] = f2bf(v);
}

// ---------------------------------------------------------------------------
// dst[e][n][k] = bf16(src[e][k][n])  -- pre-transpose weights so B-fragments
// for the WMMA 32x16 bf16 layout become two contiguous 16B chunks per lane.
__global__ void k_transpose_bf16(const float* __restrict__ src, unsigned short* __restrict__ dst,
                                 int K, int Nn, int total) {
  int idx = blockIdx.x * blockDim.x + threadIdx.x;
  if (idx >= total) return;
  int e = idx / (K * Nn);
  int r = idx - e * (K * Nn);
  int n = r / K;
  int k = r - n * K;
  dst[idx] = f2bf(src[(size_t)(e * K + k) * Nn + n]);
}

// ---------------------------------------------------------------------------
// hr = gelu(x @ Wr1 + br1) in bf16.  block: 256 thr = 8 waves, 64 tokens/block.
__global__ void __launch_bounds__(256) k_router(const unsigned short* __restrict__ xbf,
                                                const unsigned short* __restrict__ Wr1t,
                                                const float* __restrict__ br1,
                                                unsigned short* __restrict__ hrbf) {
  __shared__ alignas(16) unsigned short xs[64 * DIN];  // 40 KB
  int tid = threadIdx.x;
  int t0 = blockIdx.x * 64;
  lds_fill(xs, xbf + (size_t)t0 * DIN, 64 * DIN / 8, tid, 256);
  __syncthreads();
  int wave = tid >> 5, lane = tid & 31, m = lane & 15, half = lane >> 4;
#pragma unroll
  for (int i = 0; i < 8; ++i) {
    int tile = wave * 8 + i;          // 64 tiles: 4 m-tiles x 16 n-tiles
    int mt = tile >> 4, nt = tile & 15;
    v8f acc = {};
    const unsigned short* arow = xs + (size_t)(mt * 16 + m) * DIN;
    const unsigned short* brow = Wr1t + (size_t)(nt * 16 + m) * DIN;  // col = lane&15
#pragma unroll
    for (int k0 = 0; k0 < DIN; k0 += 32) {
      v16bf av = load16(arow + k0 + half * 8, arow + k0 + 16 + half * 8);
      v16bf bv = load16(brow + k0 + half * 8, brow + k0 + 16 + half * 8);
      acc = wmma_bf16(av, bv, acc);
    }
    int ncol = nt * 16 + m;
    float bias = br1[ncol];
#pragma unroll
    for (int j = 0; j < 8; ++j) {  // lane,j -> (M = j+8*half, N = lane&15)
      int row = t0 + mt * 16 + j + 8 * half;
      hrbf[(size_t)row * DRR + ncol] = f2bf(gelu_tanh(acc[j] + bias));
    }
  }
}

// ---------------------------------------------------------------------------
// Gate: logits = hr @ Wr2 + br2; top-2 softmax -> gate_weights, gate_logits.
__global__ void __launch_bounds__(256) k_gate(const unsigned short* __restrict__ hrbf,
                                              const float* __restrict__ Wr2,
                                              const float* __restrict__ br2,
                                              float* __restrict__ gw, float* __restrict__ gl) {
  __shared__ float sW[DRR * EE];  // 8 KB
  int tid = threadIdx.x;
  for (int i = tid; i < DRR * EE; i += 256) sW[i] = Wr2[i];
  __syncthreads();
  int t = blockIdx.x * 256 + tid;
  float lg[EE];
#pragma unroll
  for (int e = 0; e < EE; ++e) lg[e] = br2[e];
  const unsigned short* hr = hrbf + (size_t)t * DRR;
  for (int k = 0; k < DRR; ++k) {
    float h = bf2f(hr[k]);
#pragma unroll
    for (int e = 0; e < EE; ++e) lg[e] += h * sW[k * EE + e];
  }
  // top-2 (ties -> first index, matching lax.top_k)
  int i1 = 0; float v1 = lg[0];
#pragma unroll
  for (int e = 1; e < EE; ++e) if (lg[e] > v1) { v1 = lg[e]; i1 = e; }
  int i2 = (i1 == 0) ? 1 : 0; float v2 = lg[i2];
#pragma unroll
  for (int e = 0; e < EE; ++e) if (e != i1 && lg[e] > v2) { v2 = lg[e]; i2 = e; }
  float w2 = 1.0f / (1.0f + __expf(v1 - v2));  // softmax over {v1,v2}
  float w1 = 1.0f - w2;
  float g[EE];
#pragma unroll
  for (int e = 0; e < EE; ++e) g[e] = 0.0f;
  g[i1] = w1; g[i2] = w2;
#pragma unroll
  for (int e = 0; e < EE; ++e) {
    gw[(size_t)t * EE + e] = g[e];
    gl[(size_t)t * EE + e] = lg[e];  // TEMP == 1.0
  }
}

// ---------------------------------------------------------------------------
// Fused experts: per 32-token block, loop experts, gate folded into he rows.
__global__ void __launch_bounds__(256) k_experts(const unsigned short* __restrict__ xbf,
                                                 const unsigned short* __restrict__ We1t,
                                                 const unsigned short* __restrict__ We2t,
                                                 const float* __restrict__ be1,
                                                 const float* __restrict__ be2,
                                                 const float* __restrict__ gw,
                                                 float* __restrict__ out) {
  __shared__ alignas(16) unsigned short xs[32 * DIN];  // 20 KB
  __shared__ alignas(16) unsigned short hs[32 * DHH];  // 32 KB
  int tid = threadIdx.x;
  int t0 = blockIdx.x * 32;
  lds_fill(xs, xbf + (size_t)t0 * DIN, 32 * DIN / 8, tid, 256);
  __syncthreads();
  int wave = tid >> 5, lane = tid & 31, m = lane & 15, half = lane >> 4;
  int mt = wave & 1, wg = wave >> 1;  // 2 m-tiles, 4 wave-groups across N

  v8f acc[4];  // persistent output accumulators: 4 n-tiles of 16 per wave
#pragma unroll
  for (int i = 0; i < 4; ++i) acc[i] = (v8f){};

#pragma unroll 1
  for (int e = 0; e < EE; ++e) {
    // pull next expert's weight blocks toward the WGP while we compute this one
    if (e + 1 < EE) {
      __builtin_prefetch(We1t + (size_t)(e + 1) * DHH * DIN + (size_t)tid * 1280, 0, 3);
      __builtin_prefetch(We2t + (size_t)(e + 1) * DD * DHH + (size_t)tid * 512, 0, 3);
    }

    float g[8];  // gate weight per output row this wave owns
#pragma unroll
    for (int j = 0; j < 8; ++j)
      g[j] = gw[(size_t)(t0 + mt * 16 + j + 8 * half) * EE + e];

    // GEMM1: hs = g * gelu(x @ We1[e] + be1[e])   (32x512, K=320)
    const unsigned short* W1 = We1t + (size_t)e * DHH * DIN;
#pragma unroll
    for (int i = 0; i < 8; ++i) {
      int nt = wg * 8 + i;  // 32 n-tiles over 4 wave-groups
      v8f a = {};
      const unsigned short* arow = xs + (size_t)(mt * 16 + m) * DIN;
      const unsigned short* brow = W1 + (size_t)(nt * 16 + m) * DIN;
#pragma unroll
      for (int k0 = 0; k0 < DIN; k0 += 32) {
        v16bf av = load16(arow + k0 + half * 8, arow + k0 + 16 + half * 8);
        v16bf bv = load16(brow + k0 + half * 8, brow + k0 + 16 + half * 8);
        a = wmma_bf16(av, bv, a);
      }
      int ncol = nt * 16 + m;
      float bias = be1[e * DHH + ncol];
#pragma unroll
      for (int j = 0; j < 8; ++j) {
        float v = gelu_tanh(a[j] + bias) * g[j];
        hs[(size_t)(mt * 16 + j + 8 * half) * DHH + ncol] = f2bf(v);
      }
    }
    __syncthreads();

    // GEMM2: acc += hs @ We2[e]  (32x256, K=512);  + g * be2[e]
    const unsigned short* W2 = We2t + (size_t)e * DD * DHH;
#pragma unroll
    for (int i = 0; i < 4; ++i) {
      int nt = wg * 4 + i;  // 16 n-tiles over 4 wave-groups
      const unsigned short* arow = hs + (size_t)(mt * 16 + m) * DHH;
      const unsigned short* brow = W2 + (size_t)(nt * 16 + m) * DHH;
      v8f c = acc[i];
#pragma unroll
      for (int k0 = 0; k0 < DHH; k0 += 32) {
        v16bf av = load16(arow + k0 + half * 8, arow + k0 + 16 + half * 8);
        v16bf bv = load16(brow + k0 + half * 8, brow + k0 + 16 + half * 8);
        c = wmma_bf16(av, bv, c);
      }
      int ncol = nt * 16 + m;
      float b2 = be2[e * DD + ncol];
#pragma unroll
      for (int j = 0; j < 8; ++j) c[j] += g[j] * b2;
      acc[i] = c;
    }
    __syncthreads();  // hs reused next expert
  }

  // stage_delta epilogue
#pragma unroll
  for (int i = 0; i < 4; ++i) {
    int ncol = (wg * 4 + i) * 16 + m;
#pragma unroll
    for (int j = 0; j < 8; ++j)
      out[(size_t)(t0 + mt * 16 + j + 8 * half) * DD + ncol] = acc[i][j];
  }
}

// ---------------------------------------------------------------------------
extern "C" void kernel_launch(void* const* d_in, const int* in_sizes, int n_in,
                              void* d_out, int out_size, void* d_ws, size_t ws_size,
                              hipStream_t stream) {
  (void)in_sizes; (void)n_in; (void)out_size; (void)ws_size;
  const float* hidden = (const float*)d_in[0];
  const float* feat   = (const float*)d_in[1];
  const float* W_feat = (const float*)d_in[2];
  const float* b_feat = (const float*)d_in[3];
  const float* Wr1    = (const float*)d_in[4];
  const float* br1    = (const float*)d_in[5];
  const float* Wr2    = (const float*)d_in[6];
  const float* br2    = (const float*)d_in[7];
  const float* We1    = (const float*)d_in[8];
  const float* be1    = (const float*)d_in[9];
  const float* We2    = (const float*)d_in[10];
  const float* be2    = (const float*)d_in[11];

  float* out = (float*)d_out;                    // stage_delta [NTOK, DD]
  float* gwo = out + (size_t)NTOK * DD;          // gate_weights [NTOK, EE]
  float* glo = gwo + (size_t)NTOK * EE;          // gate_logits  [NTOK, EE]

  // scratch layout (bf16 as ushort)
  unsigned short* xbf  = (unsigned short*)d_ws;            // [NTOK, DIN]
  unsigned short* hrbf = xbf + (size_t)NTOK * DIN;         // [NTOK, DRR]
  unsigned short* Wr1t = hrbf + (size_t)NTOK * DRR;        // [DRR, DIN]
  unsigned short* We1t = Wr1t + (size_t)DRR * DIN;         // [EE, DHH, DIN]
  unsigned short* We2t = We1t + (size_t)EE * DHH * DIN;    // [EE, DD, DHH]

  k_build_x<<<NTOK, DIN, 0, stream>>>(hidden, feat, W_feat, b_feat, xbf);

  {
    int total = DRR * DIN;
    k_transpose_bf16<<<(total + 255) / 256, 256, 0, stream>>>(Wr1, Wr1t, DIN, DRR, total);
  }
  {
    int total = EE * DIN * DHH;
    k_transpose_bf16<<<(total + 255) / 256, 256, 0, stream>>>(We1, We1t, DIN, DHH, total);
  }
  {
    int total = EE * DHH * DD;
    k_transpose_bf16<<<(total + 255) / 256, 256, 0, stream>>>(We2, We2t, DHH, DD, total);
  }

  k_router<<<NTOK / 64, 256, 0, stream>>>(xbf, Wr1t, br1, hrbf);
  k_gate<<<NTOK / 256, 256, 0, stream>>>(hrbf, Wr2, br2, gwo, glo);
  k_experts<<<NTOK / 32, 256, 0, stream>>>(xbf, We1t, We2t, be1, be2, gwo, out);
}
